// ModulatedDeformConv2d_2628519985378
// MI455X (gfx1250) — compile-verified, matrix operand order checked by reading, not run on hardware
//
#include <hip/hip_runtime.h>

typedef __attribute__((ext_vector_type(16))) _Float16 v16h;
typedef __attribute__((ext_vector_type(8)))  _Float16 v8h;
typedef __attribute__((ext_vector_type(8)))  float    v8f;

#define B_   4
#define C_   128
#define H_   64
#define W_   64
#define COUT 128
#define KW_  3
#define K_   9
#define DG   2
#define CG   (C_/DG)        // 64 channels per deform group
#define P_   (H_*W_)        // 4096 output positions
#define CK   (C_*K_)        // 1152 GEMM K-dim

// -------------------------------------------------------------------------
// Kernel 1: weight fp32 -> fp16 with K-dim permutation:
//   w16[cout][k*128 + c] = w[cout][c*9 + k]
// (GEMM reduction order is permutation-invariant; this makes im2col stores
//  fully contiguous.)
// -------------------------------------------------------------------------
__global__ void mdcn_wcvt(const float* __restrict__ w, _Float16* __restrict__ w16) {
  int i = blockIdx.x * blockDim.x + threadIdx.x;
  if (i >= COUT * CK) return;
  int ck   = i % CK;
  int cout = i / CK;
  int k = ck / C_;
  int c = ck % C_;
  w16[i] = (_Float16)w[(size_t)cout * CK + c * K_ + k];
}

// -------------------------------------------------------------------------
// Kernel 2: deformable modulated im2col -> colT[b][p][k*128 + c] in fp16.
// One thread per (b, dg, k, p): bilinear weights computed once, reused
// across the 64 channels of the deform group. Gathers: adjacent lanes =
// adjacent spatial p -> same cachelines. Stores: 64 contiguous f16 per
// thread = 8x global_store_b128 (16B aligned).
// -------------------------------------------------------------------------
__global__ void mdcn_im2col_f16(const float* __restrict__ x,
                                const float* __restrict__ off,
                                const float* __restrict__ msk,
                                _Float16* __restrict__ colT) {
  int tid = blockIdx.x * blockDim.x + threadIdx.x;
  if (tid >= B_ * DG * K_ * P_) return;
  int p  = tid & (P_ - 1);
  int r  = tid >> 12;
  int k  = r % K_;  r /= K_;
  int dg = r & 1;
  int b  = r >> 1;
  int ho = p >> 6, wo = p & 63;

  int offBase = ((b * (DG * 2 * K_) + (dg * K_ + k) * 2) * P_) + p;
  float dy = off[offBase];
  float dx = off[offBase + P_];
  float m  = msk[((b * (DG * K_) + dg * K_ + k) * P_) + p];

  // stride=1, pad=1, dil=1
  float py = dy + (float)(ho - 1) + (float)(k / KW_);
  float px = dx + (float)(wo - 1) + (float)(k % KW_);
  float y0f = floorf(py), x0f = floorf(px);
  float ly = py - y0f, lx = px - x0f;
  int y0 = (int)y0f, x0 = (int)x0f;
  int y1 = y0 + 1,   x1 = x0 + 1;
  float vy0 = (y0 >= 0 && y0 < H_) ? 1.f : 0.f;
  float vy1 = (y1 >= 0 && y1 < H_) ? 1.f : 0.f;
  float vx0 = (x0 >= 0 && x0 < W_) ? 1.f : 0.f;
  float vx1 = (x1 >= 0 && x1 < W_) ? 1.f : 0.f;
  int cy0 = min(max(y0, 0), H_ - 1), cy1 = min(max(y1, 0), H_ - 1);
  int cx0 = min(max(x0, 0), W_ - 1), cx1 = min(max(x1, 0), W_ - 1);
  int i00 = cy0 * W_ + cx0, i01 = cy0 * W_ + cx1;
  int i10 = cy1 * W_ + cx0, i11 = cy1 * W_ + cx1;
  float w00 = (1.f - ly) * (1.f - lx) * vy0 * vx0 * m;
  float w01 = (1.f - ly) * lx         * vy0 * vx1 * m;
  float w10 = ly         * (1.f - lx) * vy1 * vx0 * m;
  float w11 = ly         * lx         * vy1 * vx1 * m;

  const float* xp = x + ((size_t)(b * C_ + dg * CG) * P_);
  // contiguous 64-channel run in the permuted K layout
  v8h* dst = (v8h*)(colT + ((size_t)b * P_ + p) * CK + (size_t)k * C_ + dg * CG);

#pragma unroll
  for (int i0 = 0; i0 < CG; i0 += 8) {
    v8h vv;
#pragma unroll
    for (int j = 0; j < 8; ++j) {
      const float* xc = xp + (size_t)(i0 + j) * P_;
      vv[j] = (_Float16)(w00 * xc[i00] + w01 * xc[i01] +
                         w10 * xc[i10] + w11 * xc[i11]);
    }
    dst[i0 / 8] = vv;   // global_store_b128
  }
}

// -------------------------------------------------------------------------
// Kernel 3: WMMA GEMM  out[b](128x4096) = W16(128x1152) * colT[b]^T
// 32x32 tile per wave (2x2 of 16x16), v_wmma_f32_16x16x32_f16, f32 acc,
// fused bias. 4 fragment loads feed 4 WMMAs per K-step.
// Fragment layout (16x32 f16): lane group g=lane/16, row M(=N)=lane%16,
//   VGPRs 0-3 <- K[kk + 8g : +8), VGPRs 4-7 <- K[kk+16+8g : +8)
// -------------------------------------------------------------------------
__global__ void __launch_bounds__(256)
mdcn_wmma_gemm(const _Float16* __restrict__ wgt,
               const _Float16* __restrict__ colT,
               const float* __restrict__ bias,
               float* __restrict__ out) {
  int wave = threadIdx.x >> 5;
  int lane = threadIdx.x & 31;
  int wid  = blockIdx.x * 8 + wave;   // 2048 waves total
  int b    = wid >> 9;                // 512 waves per batch
  int t    = wid & 511;
  int mbase = (t >> 7) << 5;          // 4 m-blocks of 32
  int pbase = (t & 127) << 5;         // 128 p-blocks of 32
  int half  = lane >> 4;
  int rr    = lane & 15;

  const _Float16* a0 = wgt  + (size_t)(mbase + rr) * CK + half * 8;
  const _Float16* a1 = a0 + (size_t)16 * CK;
  const _Float16* b0 = colT + ((size_t)b * P_ + pbase + rr) * CK + half * 8;
  const _Float16* b1 = b0 + (size_t)16 * CK;

  v8f acc00 = {}, acc01 = {}, acc10 = {}, acc11 = {};
  for (int kk = 0; kk < CK; kk += 32) {
    union { v16h v; v8h h[2]; } A0, A1, B0, B1;
    A0.h[0] = *(const v8h*)(a0 + kk); A0.h[1] = *(const v8h*)(a0 + kk + 16);
    A1.h[0] = *(const v8h*)(a1 + kk); A1.h[1] = *(const v8h*)(a1 + kk + 16);
    B0.h[0] = *(const v8h*)(b0 + kk); B0.h[1] = *(const v8h*)(b0 + kk + 16);
    B1.h[0] = *(const v8h*)(b1 + kk); B1.h[1] = *(const v8h*)(b1 + kk + 16);
    acc00 = __builtin_amdgcn_wmma_f32_16x16x32_f16(false, A0.v, false, B0.v,
                                                   (short)0, acc00, false, false);
    acc01 = __builtin_amdgcn_wmma_f32_16x16x32_f16(false, A0.v, false, B1.v,
                                                   (short)0, acc01, false, false);
    acc10 = __builtin_amdgcn_wmma_f32_16x16x32_f16(false, A1.v, false, B0.v,
                                                   (short)0, acc10, false, false);
    acc11 = __builtin_amdgcn_wmma_f32_16x16x32_f16(false, A1.v, false, B1.v,
                                                   (short)0, acc11, false, false);
  }

  // C/D layout: VGPR v -> M = v + 8*half (+ tile offset), N = lane%16 (+ tile)
  float* o00 = out + ((size_t)b * COUT + mbase + half * 8) * P_ + pbase + rr;
  float* o10 = o00 + (size_t)16 * P_;
#pragma unroll
  for (int v = 0; v < 8; ++v) {
    float bi0 = bias[mbase + half * 8 + v];
    float bi1 = bias[mbase + 16 + half * 8 + v];
    o00[(size_t)v * P_]      = acc00[v] + bi0;
    o00[(size_t)v * P_ + 16] = acc01[v] + bi0;
    o10[(size_t)v * P_]      = acc10[v] + bi1;
    o10[(size_t)v * P_ + 16] = acc11[v] + bi1;
  }
}

// -------------------------------------------------------------------------
extern "C" void kernel_launch(void* const* d_in, const int* in_sizes, int n_in,
                              void* d_out, int out_size, void* d_ws, size_t ws_size,
                              hipStream_t stream) {
  const float* x      = (const float*)d_in[0];
  const float* offset = (const float*)d_in[1];
  const float* mask   = (const float*)d_in[2];
  const float* weight = (const float*)d_in[3];
  const float* bias   = (const float*)d_in[4];
  float* out = (float*)d_out;

  // workspace: [0, 294912B) weight fp16 (permuted); then colT fp16 (~37.7MB)
  _Float16* w16  = (_Float16*)d_ws;
  _Float16* colT = w16 + (size_t)COUT * CK;

  mdcn_wcvt<<<(COUT * CK + 255) / 256, 256, 0, stream>>>(weight, w16);

  int nSamp = B_ * DG * K_ * P_;                 // 294912
  mdcn_im2col_f16<<<nSamp / 256, 256, 0, stream>>>(x, offset, mask, colT);

  int nBlocks = (B_ * (COUT / 32) * (P_ / 32)) / 8;   // 2048 waves / 8
  mdcn_wmma_gemm<<<nBlocks, 256, 0, stream>>>(w16, colT, bias, out);

  (void)in_sizes; (void)n_in; (void)out_size; (void)ws_size;
}
